// FavorAttention_68917045232094
// MI455X (gfx1250) — matrix operand order repaired and checked by reading, not compiled, not run on hardware
//
#include <hip/hip_runtime.h>
#include <math.h>

// ---------------------------------------------------------------------------
// FAVOR+ linear attention for MI455X (gfx1250, wave32, WMMA 16x16x32 f16->f32)
//
// Pipeline (stream order = dependency order):
//   k_init        : segmax=0, seg_start=N, seg_end=0
//   k_seg_bounds  : segment [start,end) via atomics over sorted batch_seg
//   k_pack_w      : omega -> B-fragment-ready f16 layout (32 KB)
//   k_gemm_u (x2) : U = (X * d^-0.25) @ omega              [WMMA, b128 loads]
//   k_row_stats(2): h, per-row max (Q) / segment max via uint-atomicMax (K)
//   k_phi (x2)    : U <- (exp(U - h - max) + eps)/sqrt(m)  (in place)
//   k_ksum        : Ksum_b = sum_{j in b} Kp_j
//   k_segS        : S_b = sum_{j in b} Kp_j^T (outer) V_j  [WMMA, LDS-staged]
//                   stores S_b as f16 in fragment-major [dv][feat] layout
//   k_norm        : norm_i = Qp_i . Ksum_b + eps
//   k_out         : out_i = (Qp_i @ S_b) / norm_i          [WMMA]
//                   S_b staged to LDS via global_load_async_to_lds_b128
// ---------------------------------------------------------------------------

typedef __attribute__((ext_vector_type(16))) _Float16 v16h;
typedef __attribute__((ext_vector_type(8)))  _Float16 v8h;
typedef __attribute__((ext_vector_type(8)))  float    v8f;

constexpr int D    = 128;   // qk dim
constexpr int M    = 128;   // feature dim (omega cols)
constexpr int DV   = 128;   // value dim
constexpr int BSEG = 64;    // number of segments (reference: num_batch = 64)

constexpr float PHI_EPS    = 1e-4f;
constexpr float NORM_EPS   = 1e-8f;
constexpr float SCALE_X    = 0.29730177875068026f;   // 128^-0.25
constexpr float H_SCALE    = 0.04419417382415922f;   // 1/(2*sqrt(128))
constexpr float INV_SQRT_M = 0.08838834764831845f;   // 1/sqrt(128)

// --- WMMA fragment index mapping (wave32, f16 16x16x32, per ISA 7.12.2) ----
// A 16x32 (MxK): lane l -> M = l&15 ; element e -> K = e + (e<8?0:8) + (l<16?0:8)
//   i.e. per lane two contiguous 8-element K-runs at (l<16?0:8) and 16+(l<16?0:8)
// B 32x16 (KxN): lane l -> N = l&15 ; element e -> K as for A (same two 8-element runs)
//   k_segS/k_out use K = e + (l<16?0:16) for both A and B instead
// C/D f32 16x16: vgpr r -> M = r + (l<16?0:8) ; N = l&15

__device__ __forceinline__ v8f wmma_f16(v16h a, v16h b, v8f c) {
  return __builtin_amdgcn_wmma_f32_16x16x32_f16(false, a, false, b,
                                                (short)0, c, false, false);
}

__device__ __forceinline__ v16h cat8(v8h a, v8h b) {
  return __builtin_shufflevector(a, b, 0, 1, 2, 3, 4, 5, 6, 7,
                                       8, 9, 10, 11, 12, 13, 14, 15);
}

// ---------------------------------------------------------------------------
__global__ void k_init(float* segmax, int* seg_start, int* seg_end, int n) {
  int t = threadIdx.x;
  if (t < BSEG) { segmax[t] = 0.0f; seg_start[t] = n; seg_end[t] = 0; }
}

__global__ void k_seg_bounds(const long long* __restrict__ seg,
                             int* seg_start, int* seg_end, int n) {
  int i = blockIdx.x * blockDim.x + threadIdx.x;
  if (i < n) {
    int b = (int)seg[i];
    atomicMin(&seg_start[b], i);
    atomicMax(&seg_end[b], i + 1);
  }
}

// Pack omega (D x M, f32 row-major) into per-(coltile,kchunk,lane) contiguous
// 16-half B fragments: Wp[((ct*4+kci)*32+lane)*16 + e] = W[k(e,lane)][ct*16+l16]
__global__ __launch_bounds__(256)
void k_pack_w(const float* __restrict__ W, _Float16* __restrict__ Wp) {
  const int t    = blockIdx.x * blockDim.x + threadIdx.x;   // 0..1023
  const int ct   = t >> 7;          // col tile 0..7
  const int kci  = (t >> 5) & 3;    // k chunk 0..3
  const int lane = t & 31;
  const bool lo  = lane < 16;
  const int  l16 = lane & 15;
  _Float16* dst  = Wp + ((size_t)(ct * 4 + kci) * 32 + lane) * 16;
  const int kc   = kci * 32;
#pragma unroll
  for (int e = 0; e < 16; ++e) {
    const int kb = kc + (e < 8 ? e : e + 8) + (lo ? 0 : 8);
    dst[e] = (_Float16)W[(size_t)kb * M + ct * 16 + l16];
  }
}

// U[row, col] = sum_k (scale*X[row,k]) * W[k,col]. Requires n % 16 == 0.
__global__ __launch_bounds__(256)
void k_gemm_u(const float* __restrict__ X, const _Float16* __restrict__ Wp,
              float* __restrict__ U, int n) {
  const int wave = threadIdx.x >> 5;
  const int lane = threadIdx.x & 31;
  const bool lo  = lane < 16;
  const int  l16 = lane & 15;
  const int  row0 = blockIdx.x * 16;
  const int  col0 = wave * 16;
  if (row0 >= n) return;

  const float* Xr = X + (size_t)(row0 + l16) * D;
  __builtin_prefetch(Xr, 0, 0);                 // -> global_prefetch_b8

  v8f acc = {};
#pragma unroll
  for (int kci = 0; kci < 4; ++kci) {
    const int kc  = kci * 32;
    const int off = kc + (lo ? 0 : 8);
    const float4 x0 = *(const float4*)(Xr + off);
    const float4 x1 = *(const float4*)(Xr + off + 4);
    const float4 x2 = *(const float4*)(Xr + off + 16);
    const float4 x3 = *(const float4*)(Xr + off + 20);
    v16h a;
    a[0]  = (_Float16)(SCALE_X * x0.x);  a[1]  = (_Float16)(SCALE_X * x0.y);
    a[2]  = (_Float16)(SCALE_X * x0.z);  a[3]  = (_Float16)(SCALE_X * x0.w);
    a[4]  = (_Float16)(SCALE_X * x1.x);  a[5]  = (_Float16)(SCALE_X * x1.y);
    a[6]  = (_Float16)(SCALE_X * x1.z);  a[7]  = (_Float16)(SCALE_X * x1.w);
    a[8]  = (_Float16)(SCALE_X * x2.x);  a[9]  = (_Float16)(SCALE_X * x2.y);
    a[10] = (_Float16)(SCALE_X * x2.z);  a[11] = (_Float16)(SCALE_X * x2.w);
    a[12] = (_Float16)(SCALE_X * x3.x);  a[13] = (_Float16)(SCALE_X * x3.y);
    a[14] = (_Float16)(SCALE_X * x3.z);  a[15] = (_Float16)(SCALE_X * x3.w);
    const v16h b = *(const v16h*)(Wp + ((size_t)(wave * 4 + kci) * 32 + lane) * 16);
    acc = wmma_f16(a, b, acc);
  }
#pragma unroll
  for (int r = 0; r < 8; ++r) {
    const int row = row0 + r + (lo ? 0 : 8);
    U[(size_t)row * M + col0 + l16] = acc[r];
  }
}

// per-row: h = sum(x^2)*H_SCALE ; rmax = max_m U[i,m]
__global__ __launch_bounds__(128)
void k_row_stats(const float* __restrict__ X, const float* __restrict__ U,
                 float* __restrict__ h, float* __restrict__ rowmax,
                 float* __restrict__ segmax, const long long* __restrict__ seg,
                 int isKey) {
  const int i = blockIdx.x;
  const int t = threadIdx.x;       // 128 threads
  __shared__ float sh[128];
  __shared__ float sm[128];
  const float x = X[(size_t)i * D + t];
  sh[t] = x * x;
  sm[t] = U[(size_t)i * M + t];
  __syncthreads();
  for (int s = 64; s > 0; s >>= 1) {
    if (t < s) { sh[t] += sh[t + s]; sm[t] = fmaxf(sm[t], sm[t + s]); }
    __syncthreads();
  }
  if (t == 0) {
    h[i] = sh[0] * H_SCALE;
    const float rm = sm[0];
    if (isKey) {
      // segmax pre-init 0.0f; only positives compete -> uint order == float order.
      if (rm > 0.0f)
        atomicMax((unsigned int*)&segmax[(int)seg[i]], __float_as_uint(rm));
    } else {
      rowmax[i] = rm;
    }
  }
}

// U <- (exp(U - h - max) + PHI_EPS) / sqrt(m)   (in place)
__global__ void k_phi(float* __restrict__ U, const float* __restrict__ h,
                      const float* __restrict__ rowmax,
                      const float* __restrict__ segmax,
                      const long long* __restrict__ seg, int isKey, int n) {
  const int idx = blockIdx.x * blockDim.x + threadIdx.x;
  if (idx >= n * M) return;
  const int i = idx / M;
  const float mx = isKey ? segmax[(int)seg[i]] : rowmax[i];
  const float v  = expf(U[idx] - h[i] - mx);
  U[idx] = (v + PHI_EPS) * INV_SQRT_M;
}

__global__ __launch_bounds__(128)
void k_ksum(const float* __restrict__ Kp, const int* __restrict__ seg_start,
            const int* __restrict__ seg_end, float* __restrict__ Ksum) {
  const int b = blockIdx.x;
  const int t = threadIdx.x;
  float s = 0.0f;
  const int e = seg_end[b];
  for (int j = seg_start[b]; j < e; ++j) s += Kp[(size_t)j * M + t];
  Ksum[(size_t)b * M + t] = s;
}

// S_b[f, d] = sum_{rows j in b} Kp[j,f] * V[j,d]
// Block b, 8 waves; wave w owns feature strip [w*16, w*16+16), all 8 dv tiles.
// Kp/V row chunks staged into LDS transposed (fragment-major), zero-padded.
// Output stored as f16 in [dv][feat] (fragment-major) layout -> ready for the
// async byte-copy LDS staging in k_out.
__global__ __launch_bounds__(256)
void k_segS(const float* __restrict__ Kp, const float* __restrict__ V,
            const int* __restrict__ seg_start, const int* __restrict__ seg_end,
            _Float16* __restrict__ Sp, int n) {
  __shared__ __align__(32) _Float16 LA[128 * 32];   // [feat][row]  8 KB
  __shared__ __align__(32) _Float16 LV[128 * 32];   // [col ][row]  8 KB
  const int tid  = threadIdx.x;
  const int b    = blockIdx.x;
  const int wave = tid >> 5;
  const int lane = tid & 31;
  const bool lo  = lane < 16;
  const int  l16 = lane & 15;
  const int start = seg_start[b];
  const int end   = seg_end[b];

  v8f acc[8] = {};
  for (int r0 = start; r0 < end; r0 += 32) {        // uniform per block
    __syncthreads();
    // Stage 32 rows x 128 cols of Kp and V, transposed, zero-padded past end.
    for (int i = tid; i < 32 * 128 / 4; i += 256) {
      const int ro  = (i * 4) >> 7;                 // row offset 0..31
      const int c0  = (i * 4) & 127;                // col 0..124 step 4
      const int row = r0 + ro;
      const int rc  = row < n ? row : n - 1;        // clamp: load always valid
      const bool ok = row < end;                    // mask value, not the load
      const float4 kv = *(const float4*)(Kp + (size_t)rc * M + c0);
      const float4 vv = *(const float4*)(V  + (size_t)rc * DV + c0);
      LA[(c0 + 0) * 32 + ro] = (_Float16)(ok ? kv.x : 0.0f);
      LA[(c0 + 1) * 32 + ro] = (_Float16)(ok ? kv.y : 0.0f);
      LA[(c0 + 2) * 32 + ro] = (_Float16)(ok ? kv.z : 0.0f);
      LA[(c0 + 3) * 32 + ro] = (_Float16)(ok ? kv.w : 0.0f);
      LV[(c0 + 0) * 32 + ro] = (_Float16)(ok ? vv.x : 0.0f);
      LV[(c0 + 1) * 32 + ro] = (_Float16)(ok ? vv.y : 0.0f);
      LV[(c0 + 2) * 32 + ro] = (_Float16)(ok ? vv.z : 0.0f);
      LV[(c0 + 3) * 32 + ro] = (_Float16)(ok ? vv.w : 0.0f);
    }
    __syncthreads();
    // A[f,k] = Kp[r0+k, w*16+f] : one contiguous 16-half run per lane
    const _Float16* pa = &LA[(wave * 16 + l16) * 32 + (lo ? 0 : 16)];
    const v16h a = cat8(*(const v8h*)pa, *(const v8h*)(pa + 8));
#pragma unroll
    for (int t = 0; t < 8; ++t) {
      // B[k,d] = V[r0+k, t*16+d] : one contiguous 16-half run per lane
      const v16h bf = *(const v16h*)&LV[(t * 16 + l16) * 32 + (lo ? 0 : 16)];
      acc[t] = wmma_f16(a, bf, acc[t]);
    }
  }
  // Store f16, fragment-major: Sp[b][d*128 + f]; per (tile, lane) the 8
  // accumulator rows are contiguous in f -> one 16-byte store each.
#pragma unroll
  for (int t = 0; t < 8; ++t) {
    const int d = t * 16 + l16;
    _Float16* dst = Sp + (size_t)b * M * DV + (size_t)d * M
                       + wave * 16 + (lo ? 0 : 8);
    v8h hv;
#pragma unroll
    for (int r = 0; r < 8; ++r) hv[r] = (_Float16)acc[t][r];
    *(v8h*)dst = hv;
  }
}

__global__ __launch_bounds__(128)
void k_norm(const float* __restrict__ Qp, const float* __restrict__ Ksum,
            const long long* __restrict__ seg, float* __restrict__ norm) {
  const int i = blockIdx.x;
  const int t = threadIdx.x;
  __shared__ float sh[128];
  const int b = (int)seg[i];
  sh[t] = Qp[(size_t)i * M + t] * Ksum[(size_t)b * M + t];
  __syncthreads();
  for (int s = 64; s > 0; s >>= 1) {
    if (t < s) sh[t] += sh[t + s];
    __syncthreads();
  }
  if (t == 0) norm[i] = sh[0] + NORM_EPS;
}

// out[i,d] = (Qp[i,:] @ S_b) / norm[i].
// S_b (f16, already fragment-major) is staged into LDS with the CDNA5 async
// copy engine: 2048 x 16B global_load_async_to_lds_b128, drained with
// s_wait_asynccnt + barrier. Qp tiles staged per 16-row iteration.
__global__ __launch_bounds__(256)
void k_out(const float* __restrict__ Qp, const _Float16* __restrict__ Sp,
           const float* __restrict__ norm, const int* __restrict__ seg_start,
           const int* __restrict__ seg_end, float* __restrict__ out, int n) {
  __shared__ __align__(32) _Float16 LS[128 * 128];  // [col][k]  32 KB
  __shared__ __align__(32) _Float16 LQ[16 * 128];   // [row][k]   4 KB
  const int tid  = threadIdx.x;
  const int b    = blockIdx.x;
  const int wave = tid >> 5;
  const int lane = tid & 31;
  const bool lo  = lane < 16;
  const int  l16 = lane & 15;
  const int start = seg_start[b];
  const int end   = seg_end[b];

  // Async byte-copy of S_b (32 KB) into LDS. Generic shared pointer's low 32
  // bits are the LDS offset (ISA 10.2: LDS aperture maps addr[31:0] directly).
  {
    const unsigned lds0 = (unsigned)(unsigned long long)(&LS[0]);
    const unsigned long long gbase =
        (unsigned long long)(const void*)(Sp + (size_t)b * M * DV);
#pragma unroll
    for (int it = 0; it < 8; ++it) {
      const unsigned off = (unsigned)(it * 256 + tid) * 16u;  // 16B chunks
      asm volatile("global_load_async_to_lds_b128 %0, %1, %2"
                   :: "v"(lds0 + off), "v"(off), "s"(gbase)
                   : "memory");
    }
    asm volatile("s_wait_asynccnt 0x0" ::: "memory");
  }
  __syncthreads();

  for (int r0 = start; r0 < end; r0 += 16) {        // uniform per block
    __syncthreads();                                // prior iter reads done
    for (int i = tid; i < 16 * 128 / 4; i += 256) {
      const int ro  = (i * 4) >> 7;
      const int c0  = (i * 4) & 127;
      const int row = r0 + ro;
      const int rc  = row < n ? row : n - 1;
      const bool ok = row < end;
      const float4 qv = *(const float4*)(Qp + (size_t)rc * M + c0);
      LQ[ro * 128 + c0 + 0] = (_Float16)(ok ? qv.x : 0.0f);
      LQ[ro * 128 + c0 + 1] = (_Float16)(ok ? qv.y : 0.0f);
      LQ[ro * 128 + c0 + 2] = (_Float16)(ok ? qv.z : 0.0f);
      LQ[ro * 128 + c0 + 3] = (_Float16)(ok ? qv.w : 0.0f);
    }
    __syncthreads();
    v8f acc = {};
#pragma unroll
    for (int kci = 0; kci < 4; ++kci) {
      const int kc = kci * 32;
      const _Float16* pa = &LQ[l16 * 128 + kc + (lo ? 0 : 16)];
      const v16h a  = cat8(*(const v8h*)pa, *(const v8h*)(pa + 8));
      const v16h bf = *(const v16h*)&LS[(wave * 16 + l16) * 128 + kc + (lo ? 0 : 16)];
      acc = wmma_f16(a, bf, acc);
    }
#pragma unroll
    for (int r = 0; r < 8; ++r) {
      const int row = r0 + r + (lo ? 0 : 8);
      if (row < end) out[(size_t)row * DV + wave * 16 + l16] = acc[r] / norm[row];
    }
  }
}

// ---------------------------------------------------------------------------
extern "C" void kernel_launch(void* const* d_in, const int* in_sizes, int n_in,
                              void* d_out, int out_size, void* d_ws, size_t ws_size,
                              hipStream_t stream) {
  const float*     Q     = (const float*)d_in[0];
  const float*     K     = (const float*)d_in[1];
  const float*     V     = (const float*)d_in[2];
  const float*     omega = (const float*)d_in[3];
  const long long* seg   = (const long long*)d_in[4];   // jnp.int64
  float*           out   = (float*)d_out;

  const int n = in_sizes[4];   // 8192 rows (multiple of 16)

  // Workspace carve-up (256B aligned), ~10.5 MB total
  char*  ws  = (char*)d_ws;
  size_t off = 0;
  auto carve = [&](size_t bytes) -> void* {
    void* p = ws + off;
    off = (off + bytes + 255) & ~(size_t)255;
    return p;
  };
  float*     UQ        = (float*)carve((size_t)n * M * sizeof(float));   // -> Qp
  float*     UK        = (float*)carve((size_t)n * M * sizeof(float));   // -> Kp
  float*     hQ        = (float*)carve((size_t)n * sizeof(float));
  float*     hK        = (float*)carve((size_t)n * sizeof(float));
  float*     rowmaxQ   = (float*)carve((size_t)n * sizeof(float));
  float*     segmax    = (float*)carve(BSEG * sizeof(float));
  int*       seg_start = (int*)carve(BSEG * sizeof(int));
  int*       seg_end   = (int*)carve(BSEG * sizeof(int));
  float*     Ksum      = (float*)carve((size_t)BSEG * M * sizeof(float));
  float*     normv     = (float*)carve((size_t)n * sizeof(float));
  _Float16*  Sp        = (_Float16*)carve((size_t)BSEG * M * DV * sizeof(_Float16));
  _Float16*  Wp        = (_Float16*)carve((size_t)8 * 4 * 32 * 16 * sizeof(_Float16));

  k_init<<<1, 64, 0, stream>>>(segmax, seg_start, seg_end, n);
  k_seg_bounds<<<(n + 255) / 256, 256, 0, stream>>>(seg, seg_start, seg_end, n);
  k_pack_w<<<4, 256, 0, stream>>>(omega, Wp);

  k_gemm_u<<<(n + 15) / 16, 256, 0, stream>>>(Q, Wp, UQ, n);
  k_gemm_u<<<(n + 15) / 16, 256, 0, stream>>>(K, Wp, UK, n);

  k_row_stats<<<n, 128, 0, stream>>>(Q, UQ, hQ, rowmaxQ, segmax, seg, 0);
  k_row_stats<<<n, 128, 0, stream>>>(K, UK, hK, rowmaxQ, segmax, seg, 1);

  k_phi<<<((size_t)n * M + 255) / 256, 256, 0, stream>>>(UQ, hQ, rowmaxQ, segmax, seg, 0, n);
  k_phi<<<((size_t)n * M + 255) / 256, 256, 0, stream>>>(UK, hK, rowmaxQ, segmax, seg, 1, n);

  k_ksum<<<BSEG, 128, 0, stream>>>(UK, seg_start, seg_end, Ksum);
  k_segS<<<BSEG, 256, 0, stream>>>(UK, V, seg_start, seg_end, Sp, n);

  k_norm<<<n, 128, 0, stream>>>(UQ, Ksum, seg, normv);
  k_out<<<BSEG, 256, 0, stream>>>(UQ, Sp, normv, seg_start, seg_end, out, n);
}
